// PropModel_Multi_21251498180712
// MI455X (gfx1250) — matrix operand order repaired and checked by reading, not compiled, not run on hardware
//
#include <hip/hip_runtime.h>
#include <math.h>

// ---------------- problem constants ----------------
#define BB    32
#define NN    1024
#define ND    8
#define FEATD 2048
#define WVD   300
#define EE    20000
#define TT    2
#define NSTEP 5
#define NDIM  8192          // NN*ND
#define QPAD  2560          // relnet contraction (50*50=2500) padded to 32
#define KWV   320           // 300 padded to multiple of 32
#define CHUNK 256           // k-chunk staged through LDS in k_prop

typedef __bf16 v8bf  __attribute__((ext_vector_type(8)));
typedef __bf16 v16bf __attribute__((ext_vector_type(16)));
typedef float  v8f   __attribute__((ext_vector_type(8)));

// ---- A-fragment loader: 16-bit A 16x32, ISA 7.12.2 layout.
// lane m = lane&15; lanes 0-15 kbase=0, lanes 16-31 kbase=8;
// halves 0..7 = row[kbase..kbase+7], halves 8..15 = row[kbase+16..kbase+23]
__device__ __forceinline__ v16bf load_frag_a(const __bf16* row, int kb) {
  v8bf lo = *reinterpret_cast<const v8bf*>(row + kb);
  v8bf hi = *reinterpret_cast<const v8bf*>(row + kb + 16);
  v16bf r;
#pragma unroll
  for (int i = 0; i < 8; ++i) { r[i] = lo[i]; r[i + 8] = hi[i]; }
  return r;
}

__device__ __forceinline__ v8f vzero8() {
  v8f z = {0.f,0.f,0.f,0.f,0.f,0.f,0.f,0.f};
  return z;
}
__device__ __forceinline__ float sigf(float x) { return 1.f / (1.f + expf(-x)); }

// LDS byte offset of a generic pointer into __shared__: for the shared aperture,
// addr[31:0] is the LDS byte address (ISA 10.2 per-aperture address calc).
__device__ __forceinline__ unsigned lds_off_of(const void* p) {
  return (unsigned)(uintptr_t)p;
}

// async global -> LDS, 16B per lane, tracked by ASYNCcnt (ISA 10 / 15.18.3)
__device__ __forceinline__ void async_copy_b128(unsigned lds_byte_off, const void* gaddr) {
  asm volatile("global_load_async_to_lds_b128 %0, %1, off"
               :: "v"(lds_byte_off), "v"(gaddr)
               : "memory");
}
__device__ __forceinline__ void wait_asynccnt0() {
  asm volatile("s_wait_asynccnt 0x0" ::: "memory");
}

// ---------------- fp32 -> bf16 convert (with K padding) ----------------
__global__ void k_cvt_pad_bf16(__bf16* __restrict__ dst, const float* __restrict__ src,
                               int dstK, int srcK, int srcLD, int srcOff, int total) {
  int id = blockIdx.x * blockDim.x + threadIdx.x;
  if (id >= total) return;
  int r = id / dstK, k = id - r * dstK;
  float v = (k < srcK) ? src[(size_t)r * srcLD + srcOff + k] : 0.f;
  dst[id] = (__bf16)v;
}

// ---------------- rel_mats -> R2T[t][k][q] bf16, q=(c,d)=c*50+d ----------------
__global__ void k_rel2(__bf16* __restrict__ R2T, const float* __restrict__ rel, int total) {
  int id = blockIdx.x * blockDim.x + threadIdx.x;
  if (id >= total) return;                       // total = TT*64*QPAD
  int q  = id % QPAD;
  int tk = id / QPAD;
  int t  = tk / 64, k = tk - t * 64;
  float v = 0.f;
  if (q < 2500) {
    int c = q / 50, d = q - c * 50;
    v = rel[((size_t)t * 50 + c) * 3200 + d * 64 + k];
  }
  R2T[id] = (__bf16)v;
}

// ---------------- embnet: embs[n,j] = wordvecs[n]·W_emb[j] + b ----------------
__global__ void k_emb(const float* __restrict__ wv, const float* __restrict__ W,
                      const float* __restrict__ b, float* __restrict__ embs) {
  int id = blockIdx.x * blockDim.x + threadIdx.x;
  if (id >= NN * 50) return;
  int n = id / 50, j = id - n * 50;
  float a = b[j];
  for (int w = 0; w < WVD; ++w) a += wv[(size_t)n * WVD + w] * W[(size_t)j * WVD + w];
  embs[id] = a;
}

// ---------------- generic WMMA GEMM:  C[m,n] = sum_k A[m,k]*B[n,k] ----------------
// one wave per 16x16 tile; grid*8 waves == (M/16)*(N/16) exactly
__global__ void __launch_bounds__(256) k_gemm_bt(
    const __bf16* __restrict__ A, const __bf16* __restrict__ B, float* __restrict__ C,
    int ntilesN, int K, int ldA, int ldB, int ldC) {
  const int lane  = threadIdx.x & 31;
  const int gw    = blockIdx.x * 8 + (threadIdx.x >> 5);
  const int mtile = gw / ntilesN, ntile = gw - mtile * ntilesN;
  const int n     = lane & 15;
  const int kb    = (lane & 16) ? 8 : 0;
  const int kbB   = (lane & 16) ? 16 : 0;
  const __bf16* ar = A + (size_t)(mtile * 16 + n) * ldA;
  const __bf16* br = B + (size_t)(ntile * 16 + n) * ldB;
  v8f acc = vzero8();
  for (int kk = 0; kk < K; kk += 32) {
    v16bf a = load_frag_a(ar + kk, kb);
    v16bf b = load_frag_a(br + kk, kb);
    acc = __builtin_amdgcn_wmma_f32_16x16x32_bf16(false, a, false, b, (short)0, acc,
                                                  false, false);
  }
  const int rofs = (lane & 16) ? 8 : 0;
#pragma unroll
  for (int r = 0; r < 8; ++r)
    C[(size_t)(mtile * 16 + r + rofs) * ldC + ntile * 16 + n] = acc[r];
}

// ---------------- hidden init: relu(t_in[b]+t_wv[n]+b1) @ W2^T + b2 ----------------
__global__ void __launch_bounds__(256) k_hidden_init(
    const float* __restrict__ t_in, const float* __restrict__ t_wv,
    const float* __restrict__ b1, const float* __restrict__ W2,
    const float* __restrict__ b2, float* __restrict__ hidden, float* __restrict__ states) {
  __shared__ float lw2[8 * 512];
  __shared__ float lb1[512];
  for (int i = threadIdx.x; i < 4096; i += 256) lw2[i] = W2[i];
  for (int i = threadIdx.x; i < 512;  i += 256) lb1[i] = b1[i];
  __syncthreads();
  int bn = blockIdx.x * 256 + threadIdx.x;       // 0..32767
  int b = bn >> 10, n = bn & 1023;
  const float* ti = t_in + b * 512;
  const float* tw = t_wv + n * 512;
  float acc[8] = {0.f,0.f,0.f,0.f,0.f,0.f,0.f,0.f};
  for (int o = 0; o < 512; ++o) {
    float a = fmaxf(ti[o] + tw[o] + lb1[o], 0.f);
#pragma unroll
    for (int j = 0; j < 8; ++j) acc[j] += a * lw2[j * 512 + o];
  }
#pragma unroll
  for (int j = 0; j < 8; ++j) {
    float hv = acc[j] + b2[j];
    hidden[(size_t)b * NDIM + n * ND + j] = hv;
    states[((size_t)b * (NSTEP + 1) + 0) * NDIM + n * ND + j] = hv;
  }
}

// ---------------- relnet: fused (ue x ve) @ rel2  via WMMA, A generated on the fly --
// one wave per 16-edge tile; weights[(t*E+e)*64 + k]
__global__ void __launch_bounds__(32) k_relweights(
    const float* __restrict__ embs, const int* __restrict__ edge_u,
    const int* __restrict__ edge_v, const __bf16* __restrict__ R2T,
    float* __restrict__ wout) {
  __shared__ float ue_l[16 * 52];
  __shared__ float ve_l[16 * 52];
  const int tile = blockIdx.x;                   // 0..TT*1250-1
  const int t  = tile / (EE / 16);
  const int e0 = (tile - t * (EE / 16)) * 16;
  const int lane = threadIdx.x;
  for (int idx = lane; idx < 16 * 50; idx += 32) {
    int el = idx / 50, c = idx - el * 50;
    int eu = edge_u[t * EE + e0 + el];
    int ev = edge_v[t * EE + e0 + el];
    ue_l[el * 52 + c] = embs[eu * 50 + c];
    ve_l[el * 52 + c] = embs[ev * 50 + c];
  }
  for (int idx = lane; idx < 32; idx += 32) {    // zero the 2 pad columns
    int el = idx >> 1, c = 50 + (idx & 1);
    ue_l[el * 52 + c] = 0.f;
    ve_l[el * 52 + c] = 0.f;
  }
  __syncthreads();
  const int m   = lane & 15;
  const int kb  = (lane & 16) ? 8 : 0;
  const int kbB = (lane & 16) ? 16 : 0;
  const float* ur = &ue_l[m * 52];
  const float* vr = &ve_l[m * 52];
  v8f acc[4] = {vzero8(), vzero8(), vzero8(), vzero8()};
  for (int k0 = 0; k0 < QPAD; k0 += 32) {
    v16bf a;
#pragma unroll
    for (int h = 0; h < 16; ++h) {
      int q = k0 + ((h < 8) ? (kb + h) : (16 + kb + (h - 8)));
      int c = q / 50;
      int d = q - c * 50;                        // c<=51 stays inside padded row
      float v = ur[c] * vr[d];
      a[h] = (__bf16)((q < 2500) ? v : 0.f);     // VALU select, no EXEC change
    }
#pragma unroll
    for (int nt = 0; nt < 4; ++nt) {
      const __bf16* br =
          R2T + (size_t)(t * 64 + nt * 16 + (lane & 15)) * QPAD + k0;
      v16bf b = load_frag_a(br, kb);
      acc[nt] = __builtin_amdgcn_wmma_f32_16x16x32_bf16(false, a, false, b, (short)0,
                                                        acc[nt], false, false);
    }
  }
  const int rofs = (lane & 16) ? 8 : 0;
#pragma unroll
  for (int nt = 0; nt < 4; ++nt)
#pragma unroll
    for (int r = 0; r < 8; ++r)
      wout[(size_t)(t * EE + e0 + r + rofs) * 64 + nt * 16 + (lane & 15)] = acc[nt][r];
}

// ---------------- P construction ----------------
__global__ void k_zero16(uint4* __restrict__ p, size_t count) {
  size_t id = (size_t)blockIdx.x * blockDim.x + threadIdx.x;
  if (id < count) p[id] = make_uint4(0u, 0u, 0u, 0u);
}

__global__ void k_scatter(__bf16* __restrict__ P, const int* __restrict__ mu,
                          const int* __restrict__ mv, const float* __restrict__ w,
                          int total) {
  int id = blockIdx.x * blockDim.x + threadIdx.x;
  if (id >= total) return;
  P[(size_t)mu[id] * NDIM + mv[id]] = (__bf16)w[id];
}

__global__ void __launch_bounds__(256) k_rownorm(const __bf16* __restrict__ P,
                                                 float* __restrict__ inv) {
  __shared__ float red[256];
  const __bf16* row = P + (size_t)blockIdx.x * NDIM;
  float s = 0.f;
  for (int k = threadIdx.x; k < NDIM; k += 256) {
    float f = (float)row[k];
    s += f * f;
  }
  red[threadIdx.x] = s;
  __syncthreads();
  for (int st = 128; st > 0; st >>= 1) {
    if (threadIdx.x < st) red[threadIdx.x] += red[threadIdx.x + st];
    __syncthreads();
  }
  if (threadIdx.x == 0) {
    float nrm = sqrtf(red[0]);
    if (nrm == 0.f) nrm = 1.f;                   // norm += (norm==0)
    inv[blockIdx.x] = 1.f / nrm;
  }
}

// faithful quirk: P[i,j] /= norm_of_row_j  (column scaled by row-j norm)
__global__ void k_colscale(__bf16* __restrict__ P, const float* __restrict__ inv) {
  size_t id = (size_t)blockIdx.x * blockDim.x + threadIdx.x;
  if (id >= (size_t)NDIM * NDIM) return;
  int j = (int)(id & (NDIM - 1));
  P[id] = (__bf16)((float)P[id] * inv[j]);
}

// ---------------- propagation: x = tanh(hidden @ P^T) ----------------
// A (hidden, shared by all 8 waves of the block) is staged global->LDS with
// GLOBAL_LOAD_ASYNC_TO_LDS_B128, double-buffered, overlapped with WMMA on the
// previous chunk. B (per-wave-unique P rows) stays on the direct global path.
__global__ void __launch_bounds__(256) k_prop(const __bf16* __restrict__ hbf,
                                              const __bf16* __restrict__ P,
                                              float* __restrict__ x) {
  __shared__ __bf16 As[2][32 * CHUNK];           // 2 x 16KB double buffer
  const int tid   = threadIdx.x;
  const int lane  = tid & 31;
  const int ntile = blockIdx.x * 8 + (tid >> 5); // 0..511
  const int n   = lane & 15;
  const int kb  = (lane & 16) ? 8 : 0;
  const int kbB = (lane & 16) ? 16 : 0;
  const __bf16* br = P + (size_t)(ntile * 16 + n) * NDIM;
  v8f acc0 = vzero8(), acc1 = vzero8();

  // cooperative async stage of one 32 x CHUNK bf16 slab of hidden into As[buf]
  auto stage = [&](int chunk, int buf) {
    const __bf16* src = hbf + chunk * CHUNK;
    __bf16* dst = &As[buf][0];
    const int nvec = 32 * (CHUNK / 8);           // 16B vectors in the slab
#pragma unroll
    for (int e = tid; e < nvec; e += 256) {
      int row = e / (CHUNK / 8);
      int c8  = e - row * (CHUNK / 8);
      async_copy_b128(lds_off_of(dst + row * CHUNK + c8 * 8),
                      src + (size_t)row * NDIM + c8 * 8);
    }
  };

  stage(0, 0);
  int buf = 0;
  for (int c = 0; c < NDIM / CHUNK; ++c) {
    wait_asynccnt0();                            // my staged writes landed in LDS
    __syncthreads();                             // everyone's staged writes visible
    if (c + 1 < NDIM / CHUNK) stage(c + 1, buf ^ 1);
    const __bf16* a0base = &As[buf][n * CHUNK];        // batch rows 0..15
    const __bf16* a1base = &As[buf][(n + 16) * CHUNK]; // batch rows 16..31
    for (int kk = 0; kk < CHUNK; kk += 32) {
      __builtin_prefetch(br + c * CHUNK + kk + 512, 0, 1);  // global_prefetch_b8
      v16bf a0 = load_frag_a(a0base + kk, kb);   // ds_load_b128 pairs
      v16bf a1 = load_frag_a(a1base + kk, kb);
      v16bf b  = load_frag_a(br + c * CHUNK + kk, kb);
      acc0 = __builtin_amdgcn_wmma_f32_16x16x32_bf16(false, a0, false, b, (short)0,
                                                     acc0, false, false);
      acc1 = __builtin_amdgcn_wmma_f32_16x16x32_bf16(false, a1, false, b, (short)0,
                                                     acc1, false, false);
    }
    __syncthreads();                             // done reading As[buf]
    buf ^= 1;
  }

  const int rofs = (lane & 16) ? 8 : 0;
  const int i = ntile * 16 + n;
#pragma unroll
  for (int r = 0; r < 8; ++r) {
    x[(size_t)(r + rofs) * NDIM + i]      = tanhf(acc0[r]);
    x[(size_t)(r + rofs + 16) * NDIM + i] = tanhf(acc1[r]);
  }
}

// ---------------- GRU cell on [B*N, 8] rows ----------------
__global__ void __launch_bounds__(256) k_gru(
    const float* __restrict__ x, float* __restrict__ hidden, float* __restrict__ states,
    const float* __restrict__ W_ih, const float* __restrict__ W_hh,
    const float* __restrict__ b_ih, const float* __restrict__ b_hh, int step) {
  __shared__ float wih[192], whh[192], bih[24], bhh[24];
  if (threadIdx.x < 192) { wih[threadIdx.x] = W_ih[threadIdx.x]; whh[threadIdx.x] = W_hh[threadIdx.x]; }
  if (threadIdx.x < 24)  { bih[threadIdx.x] = b_ih[threadIdx.x]; bhh[threadIdx.x] = b_hh[threadIdx.x]; }
  __syncthreads();
  int bn = blockIdx.x * 256 + threadIdx.x;
  int b = bn >> 10, n = bn & 1023;
  float xv[8], hv[8];
#pragma unroll
  for (int k = 0; k < 8; ++k) {
    xv[k] = x[(size_t)bn * 8 + k];
    hv[k] = hidden[(size_t)bn * 8 + k];
  }
  float gi[24], gh[24];
#pragma unroll
  for (int j = 0; j < 24; ++j) {
    float si = bih[j], sh = bhh[j];
#pragma unroll
    for (int k = 0; k < 8; ++k) {
      si += xv[k] * wih[j * 8 + k];
      sh += hv[k] * whh[j * 8 + k];
    }
    gi[j] = si; gh[j] = sh;
  }
#pragma unroll
  for (int j = 0; j < 8; ++j) {
    float r  = sigf(gi[j] + gh[j]);
    float z  = sigf(gi[8 + j] + gh[8 + j]);
    float nn = tanhf(gi[16 + j] + r * gh[16 + j]);
    float hnew = (1.f - z) * nn + z * hv[j];
    hidden[(size_t)bn * 8 + j] = hnew;
    states[((size_t)b * (NSTEP + 1) + step + 1) * NDIM + n * 8 + j] = hnew;
  }
}

// ---------------- outnet ----------------
__global__ void k_out(const float* __restrict__ states, const float* __restrict__ W1,
                      const float* __restrict__ b1, const float* __restrict__ W2,
                      const float* __restrict__ b2, float* __restrict__ out) {
  int id = blockIdx.x * blockDim.x + threadIdx.x;          // < B*(S+1)*N
  if (id >= BB * (NSTEP + 1) * NN) return;
  int b   = id / ((NSTEP + 1) * NN);
  int rem = id - b * (NSTEP + 1) * NN;
  int s   = rem / NN;
  int n   = rem - s * NN;
  float st[8];
#pragma unroll
  for (int k = 0; k < 8; ++k)
    st[k] = states[((size_t)b * (NSTEP + 1) + s) * NDIM + n * 8 + k];
  float o = b2[0];
  for (int j = 0; j < 20; ++j) {
    float a = b1[j];
#pragma unroll
    for (int k = 0; k < 8; ++k) a += st[k] * W1[j * 8 + k];
    o += fmaxf(a, 0.f) * W2[j];
  }
  out[((size_t)b * NN + n) * (NSTEP + 1) + s] = o;
}

// ---------------- host orchestration ----------------
extern "C" void kernel_launch(void* const* d_in, const int* in_sizes, int n_in,
                              void* d_out, int out_size, void* d_ws, size_t ws_size,
                              hipStream_t stream) {
  (void)in_sizes; (void)n_in; (void)out_size; (void)ws_size;
  const float* inputs   = (const float*)d_in[0];
  const float* wordvecs = (const float*)d_in[1];
  const int*   edge_u   = (const int*)d_in[2];
  const int*   edge_v   = (const int*)d_in[3];
  const int*   mat_u    = (const int*)d_in[4];
  const int*   mat_v    = (const int*)d_in[5];
  const float* W_in1    = (const float*)d_in[6];
  const float* b_in1    = (const float*)d_in[7];
  const float* W_in2    = (const float*)d_in[8];
  const float* b_in2    = (const float*)d_in[9];
  const float* W_emb    = (const float*)d_in[10];
  const float* b_emb    = (const float*)d_in[11];
  const float* rel      = (const float*)d_in[12];
  const float* W_ih     = (const float*)d_in[13];
  const float* W_hh     = (const float*)d_in[14];
  const float* b_ih     = (const float*)d_in[15];
  const float* b_hh     = (const float*)d_in[16];
  const float* W_o1     = (const float*)d_in[17];
  const float* b_o1     = (const float*)d_in[18];
  const float* W_o2     = (const float*)d_in[19];
  const float* b_o2     = (const float*)d_in[20];

  char* base = (char*)d_ws;
  size_t off = 0;
  auto alloc = [&](size_t bytes) -> void* {
    void* p = base + off;
    off = (off + bytes + 255) & ~(size_t)255;
    return p;
  };
  __bf16* P     = (__bf16*)alloc((size_t)NDIM * NDIM * 2);        // 128 MB, fits L2
  float*  wbuf  = (float*) alloc((size_t)TT * EE * 64 * 4);
  __bf16* in_bf = (__bf16*)alloc((size_t)BB * FEATD * 2);
  __bf16* wv_bf = (__bf16*)alloc((size_t)NN * KWV * 2);
  __bf16* w1a   = (__bf16*)alloc((size_t)512 * FEATD * 2);
  __bf16* w1b   = (__bf16*)alloc((size_t)512 * KWV * 2);
  __bf16* r2t   = (__bf16*)alloc((size_t)TT * 64 * QPAD * 2);
  float*  t_in  = (float*) alloc((size_t)BB * 512 * 4);
  float*  t_wv  = (float*) alloc((size_t)NN * 512 * 4);
  float*  embs  = (float*) alloc((size_t)NN * 50 * 4);
  float*  invn  = (float*) alloc((size_t)NDIM * 4);
  float*  hidden= (float*) alloc((size_t)BB * NDIM * 4);
  __bf16* h_bf  = (__bf16*)alloc((size_t)BB * NDIM * 2);
  float*  xbuf  = (float*) alloc((size_t)BB * NDIM * 4);
  float*  states= (float*) alloc((size_t)BB * (NSTEP + 1) * NDIM * 4);

  // --- precision conversion / repacking ---
  k_cvt_pad_bf16<<<(BB * FEATD + 255) / 256, 256, 0, stream>>>(
      in_bf, inputs, FEATD, FEATD, FEATD, 0, BB * FEATD);
  k_cvt_pad_bf16<<<(512 * FEATD + 255) / 256, 256, 0, stream>>>(
      w1a, W_in1, FEATD, FEATD, FEATD + WVD, 0, 512 * FEATD);
  k_cvt_pad_bf16<<<(NN * KWV + 255) / 256, 256, 0, stream>>>(
      wv_bf, wordvecs, KWV, WVD, WVD, 0, NN * KWV);
  k_cvt_pad_bf16<<<(512 * KWV + 255) / 256, 256, 0, stream>>>(
      w1b, W_in1, KWV, WVD, FEATD + WVD, FEATD, 512 * KWV);
  k_rel2<<<(TT * 64 * QPAD + 255) / 256, 256, 0, stream>>>(r2t, rel, TT * 64 * QPAD);
  k_emb<<<(NN * 50 + 255) / 256, 256, 0, stream>>>(wordvecs, W_emb, b_emb, embs);

  // --- inputnet, decomposed (broadcast-concat factorization) ---
  k_gemm_bt<<<(2 * 32) / 8, 256, 0, stream>>>(in_bf, w1a, t_in, 32, FEATD, FEATD, FEATD, 512);
  k_gemm_bt<<<(64 * 32) / 8, 256, 0, stream>>>(wv_bf, w1b, t_wv, 32, KWV, KWV, KWV, 512);
  k_hidden_init<<<BB * NN / 256, 256, 0, stream>>>(t_in, t_wv, b_in1, W_in2, b_in2,
                                                   hidden, states);

  // --- relnet fused GEMM (ue x ve generated in-register) ---
  k_relweights<<<TT * (EE / 16), 32, 0, stream>>>(embs, edge_u, edge_v, r2t, wbuf);

  // --- dense propagation matrix (bf16) ---
  k_zero16<<<(int)((size_t)NDIM * NDIM * 2 / 16 / 256), 256, 0, stream>>>(
      (uint4*)P, (size_t)NDIM * NDIM * 2 / 16);
  k_scatter<<<(TT * EE * 64 + 255) / 256, 256, 0, stream>>>(P, mat_u, mat_v, wbuf,
                                                            TT * EE * 64);
  k_rownorm<<<NDIM, 256, 0, stream>>>(P, invn);
  k_colscale<<<(int)((size_t)NDIM * NDIM / 256), 256, 0, stream>>>(P, invn);

  // --- 5 propagation + GRU steps ---
  for (int s = 0; s < NSTEP; ++s) {
    k_cvt_pad_bf16<<<(BB * NDIM + 255) / 256, 256, 0, stream>>>(
        h_bf, hidden, NDIM, NDIM, NDIM, 0, BB * NDIM);
    k_prop<<<64, 256, 0, stream>>>(h_bf, P, xbuf);
    k_gru<<<BB * NN / 256, 256, 0, stream>>>(xbuf, hidden, states, W_ih, W_hh, b_ih,
                                             b_hh, s);
  }

  // --- outnet ---
  k_out<<<(BB * (NSTEP + 1) * NN + 255) / 256, 256, 0, stream>>>(
      states, W_o1, b_o1, W_o2, b_o2, (float*)d_out);
}
